// HierarchyGCN_62500364091988
// MI455X (gfx1250) — matrix operand order repaired and checked by reading, not compiled, not run on hardware
//
#include <hip/hip_runtime.h>

typedef __attribute__((ext_vector_type(16))) __bf16 v16bf;
typedef __attribute__((ext_vector_type(8)))  float  v8f;
typedef __attribute__((ext_vector_type(8)))  unsigned short ushort8;

#define N_NODES 128
#define DDIM    256
#define LDSN    136   // padded row length for hT (272B = 68 dwords -> 4-bank rotation)

__device__ __forceinline__ unsigned short f2bf(float f) {
    unsigned int u = __float_as_uint(f);
    unsigned int r = u + 0x7FFFu + ((u >> 16) & 1u);   // round-to-nearest-even
    return (unsigned short)(r >> 16);
}
__device__ __forceinline__ float bf2f(unsigned short s) {
    return __uint_as_float(((unsigned int)s) << 16);
}

union Cvt16 { ushort8 u[2]; v16bf v; };

// ---------------------------------------------------------------------------
// Prep 1: masked adjacencies in bf16, pre-swizzled into the WMMA A-operand
// register layout:  pack[((s*8+mt)*4+kt)*32 + lane][j]  with
//   m = mt*16 + lane%16,  K = kt*32 + 16*(lane/16) + j
// s=0: origin(adj_in)*adj_in     s=1: origin(adj_in)^T * adj_out
// ---------------------------------------------------------------------------
__global__ void gcn_prep_adj(const float* __restrict__ adj_in,
                             const float* __restrict__ adj_out,
                             unsigned short* __restrict__ pack) {
    int idx  = blockIdx.x * blockDim.x + threadIdx.x;   // 0..32767
    int j    = idx & 15;
    int lane = (idx >> 4) & 31;
    int kt   = (idx >> 9) & 3;
    int mt   = (idx >> 11) & 7;
    int s    = idx >> 14;
    int m = mt * 16 + (lane & 15);
    int k = kt * 32 + ((lane >> 4) << 4) + j;
    float val;
    if (s == 0) {
        float a = adj_in[m * N_NODES + k];
        float o = (a <= 0.f) ? a : 1.f;
        val = o * a;
    } else {
        float at = adj_in[k * N_NODES + m];          // origin.T
        float o  = (at <= 0.f) ? at : 1.f;
        val = o * adj_out[m * N_NODES + k];
    }
    pack[idx] = f2bf(val);
}

// ---------------------------------------------------------------------------
// Prep 2: edge biases pre-swizzled into the WMMA C/D accumulator layout:
//   ebp[(((s*8+mt)*16+dt)*32 + lane)*8 + r] = bias_s[m*DDIM + d]
//   with m = mt*16 + 8*(lane/16) + r, d = dt*16 + (lane%16)
// ---------------------------------------------------------------------------
__global__ void gcn_prep_eb(const float* __restrict__ edge_bias,
                            const float* __restrict__ out_edge_bias,
                            float* __restrict__ ebp) {
    int idx  = blockIdx.x * blockDim.x + threadIdx.x;   // 0..65535
    int r    = idx & 7;
    int lane = (idx >> 3) & 31;
    int dt   = (idx >> 8) & 15;
    int mt   = (idx >> 12) & 7;
    int s    = idx >> 15;
    int m = mt * 16 + ((lane >> 4) << 3) + r;
    int d = dt * 16 + (lane & 15);
    const float* src = (s == 0) ? edge_bias : out_edge_bias;
    ebp[idx] = src[m * DDIM + d];
}

// ---------------------------------------------------------------------------
// Main: one workgroup (8 wave32) per batch element.
// ---------------------------------------------------------------------------
__global__ void __launch_bounds__(256)
gcn_main(const float* __restrict__ h,
         const unsigned short* __restrict__ apack,
         const float* __restrict__ ebp,
         const float* __restrict__ gate_w,
         const float* __restrict__ bias_gate,
         const float* __restrict__ out_gate_w,
         const float* __restrict__ out_bias_gate,
         const float* __restrict__ loop_gate,
         float* __restrict__ out) {
    __shared__ unsigned short hT[DDIM * LDSN];   // h[b] transposed, bf16 (68KB)
    __shared__ float gw[3 * DDIM];               // gate weights
    __shared__ float gb[2 * N_NODES];            // gate biases
    __shared__ float red[3 * 256];               // reduction scratch
    __shared__ float gate[3 * N_NODES];          // sigmoid(gates)

    const int tid = threadIdx.x;
    const int b   = blockIdx.x;
    const float* hb = h + (size_t)b * (N_NODES * DDIM);

    // ---- stage h[b] -> bf16 transposed in LDS (b128 coalesced reads) ----
    for (int i = tid * 4; i < N_NODES * DDIM; i += 256 * 4) {
        float4 v = *(const float4*)(hb + i);
        int n = i >> 8;          // node
        int d = i & (DDIM - 1);  // feature
        hT[(d + 0) * LDSN + n] = f2bf(v.x);
        hT[(d + 1) * LDSN + n] = f2bf(v.y);
        hT[(d + 2) * LDSN + n] = f2bf(v.z);
        hT[(d + 3) * LDSN + n] = f2bf(v.w);
    }
    if (tid < DDIM) {
        gw[tid]            = gate_w[tid];
        gw[DDIM + tid]     = out_gate_w[tid];
        gw[2 * DDIM + tid] = loop_gate[tid];
    }
    if (tid < N_NODES) {
        gb[tid]           = bias_gate[tid];
        gb[N_NODES + tid] = out_bias_gate[tid];
    }
    __syncthreads();

    // ---- gates: two threads per node, each half of D ----
    {
        int n  = tid >> 1;
        int d0 = (tid & 1) << 7;
        float s0 = 0.f, s1 = 0.f, s2 = 0.f;
        for (int dd = 0; dd < 128; ++dd) {
            int d = d0 + dd;
            float hv = bf2f(hT[d * LDSN + n]);
            s0 += hv * gw[d];
            s1 += hv * gw[DDIM + d];
            s2 += hv * gw[2 * DDIM + d];
        }
        red[tid] = s0; red[256 + tid] = s1; red[512 + tid] = s2;
    }
    __syncthreads();
    if (tid < N_NODES) {
        float gi = red[2 * tid] + red[2 * tid + 1] + gb[tid];
        float go = red[256 + 2 * tid] + red[256 + 2 * tid + 1] + gb[N_NODES + tid];
        float gl = red[512 + 2 * tid] + red[512 + 2 * tid + 1];
        gate[tid]               = 1.f / (1.f + __expf(-gi));
        gate[N_NODES + tid]     = 1.f / (1.f + __expf(-go));
        gate[2 * N_NODES + tid] = 1.f / (1.f + __expf(-gl));
    }
    __syncthreads();

    // ---- WMMA: wave w owns m-tile w; loop 16 d-tiles x 4 k-steps ----
    const int wave   = tid >> 5;
    const int lane   = tid & 31;
    const int laneHi = lane >> 4;
    const int lane16 = lane & 15;

    v16bf aIn[4], aOut[4];
#pragma unroll
    for (int kt = 0; kt < 4; ++kt) {
        const ushort8* p0 =
            (const ushort8*)(apack + (size_t)(((0 * 8 + wave) * 4 + kt) * 32 + lane) * 16);
        Cvt16 c0; c0.u[0] = p0[0]; c0.u[1] = p0[1];
        aIn[kt] = c0.v;
        const ushort8* p1 =
            (const ushort8*)(apack + (size_t)(((1 * 8 + wave) * 4 + kt) * 32 + lane) * 16);
        Cvt16 c1; c1.u[0] = p1[0]; c1.u[1] = p1[1];
        aOut[kt] = c1.v;
    }

    // loop-invariant gate values for this lane's 8 output rows
    float gIn[8], gOut[8], gLp[8];
#pragma unroll
    for (int r = 0; r < 8; ++r) {
        int m = wave * 16 + laneHi * 8 + r;
        gIn[r]  = gate[m];
        gOut[r] = gate[N_NODES + m];
        gLp[r]  = gate[2 * N_NODES + m];
    }

    const float4* ebIn  = (const float4*)ebp + (size_t)(0 * 8 + wave) * 16 * 32 * 2;
    const float4* ebOut = (const float4*)ebp + (size_t)(1 * 8 + wave) * 16 * 32 * 2;
    float* outb = out + (size_t)b * (N_NODES * DDIM);

    for (int dt = 0; dt < 16; ++dt) {
        const int d = dt * 16 + lane16;
        v8f accIn  = {0.f, 0.f, 0.f, 0.f, 0.f, 0.f, 0.f, 0.f};
        v8f accOut = {0.f, 0.f, 0.f, 0.f, 0.f, 0.f, 0.f, 0.f};
#pragma unroll
        for (int kt = 0; kt < 4; ++kt) {
            const int nb = kt * 32 + laneHi * 16;
            const ushort8* bp = (const ushort8*)(&hT[d * LDSN + nb]);
            Cvt16 cb; cb.u[0] = bp[0]; cb.u[1] = bp[1];
            accIn = __builtin_amdgcn_wmma_f32_16x16x32_bf16(
                false, aIn[kt], false, cb.v, (short)0, accIn, false, false);
            accOut = __builtin_amdgcn_wmma_f32_16x16x32_bf16(
                false, aOut[kt], false, cb.v, (short)0, accOut, false, false);
        }
        // pre-swizzled bias strips: two b128 loads per array
        const int ei = (dt * 32 + lane) * 2;
        float4 i0 = ebIn[ei],  i1 = ebIn[ei + 1];
        float4 o0 = ebOut[ei], o1 = ebOut[ei + 1];
        float ein[8] = {i0.x, i0.y, i0.z, i0.w, i1.x, i1.y, i1.z, i1.w};
        float eou[8] = {o0.x, o0.y, o0.z, o0.w, o1.x, o1.y, o1.z, o1.w};

        // epilogue: D layout -> VGPR r: M = r + 8*laneHi, N = lane16
#pragma unroll
        for (int r = 0; r < 8; ++r) {
            int m = wave * 16 + laneHi * 8 + r;
            float hv = bf2f(hT[d * LDSN + m]);
            float v = (accIn[r] + ein[r]) * gIn[r]
                    + (accOut[r] + eou[r]) * gOut[r]
                    + hv * gLp[r];
            outb[m * DDIM + d] = fmaxf(v, 0.f);
        }
    }
}

extern "C" void kernel_launch(void* const* d_in, const int* in_sizes, int n_in,
                              void* d_out, int out_size, void* d_ws, size_t ws_size,
                              hipStream_t stream) {
    const float* label         = (const float*)d_in[0];
    const float* adj_in        = (const float*)d_in[1];
    const float* adj_out       = (const float*)d_in[2];
    const float* edge_bias     = (const float*)d_in[3];
    const float* gate_weight   = (const float*)d_in[4];
    const float* bias_gate     = (const float*)d_in[5];
    const float* out_edge_bias = (const float*)d_in[6];
    const float* out_gate_w    = (const float*)d_in[7];
    const float* out_bias_gate = (const float*)d_in[8];
    const float* loop_gate     = (const float*)d_in[9];

    unsigned short* apack = (unsigned short*)d_ws;               // 64 KB
    float* ebp = (float*)((char*)d_ws + 64 * 1024);              // 256 KB

    gcn_prep_adj<<<128, 256, 0, stream>>>(adj_in, adj_out, apack);
    gcn_prep_eb<<<256, 256, 0, stream>>>(edge_bias, out_edge_bias, ebp);
    gcn_main<<<4096, 256, 0, stream>>>(label, apack, ebp, gate_weight,
                                       bias_gate, out_gate_w, out_bias_gate,
                                       loop_gate, (float*)d_out);
}
